// MetricSwiGLU_90881507983746
// MI455X (gfx1250) — matrix operand order repaired and checked by reading, not compiled, not run on hardware
//
#include <hip/hip_runtime.h>
#include <hip/hip_bf16.h>

// ---------------------------------------------------------------------------
// MetricSwiGLU for MI455X (gfx1250): bf16 WMMA GEMMs with async-LDS double
// buffering, fused SwiGLU epilogue, metric-conv offsets, deformable depthwise
// conv, bf16 WMMA down-proj.
// ---------------------------------------------------------------------------

#define BB    2
#define LL    1024
#define DD    1024
#define INTER 2816
#define SS    32

// LDS tile geometry: one 32-K row = 64B, padded to 80B stride (20-bank stride
// => the 16 lanes of a ds_load_b128 fragment read hit disjoint bank quads).
#define ROWB     80
#define TILE_A   (128 * ROWB)          // 10240 B
#define STAGEB   (3 * TILE_A)          // 30720 B per buffer (both GEMMs)

typedef __attribute__((ext_vector_type(16))) __bf16 v16bf;
typedef __attribute__((ext_vector_type(8)))  __bf16 v8bf;
typedef __attribute__((ext_vector_type(4)))  __bf16 v4bf;
typedef __attribute__((ext_vector_type(8)))  float  v8f;

// ---------------------------------------------------------------------------
// gfx1250 async global->LDS copy (ASYNCcnt path) + wait.
// Builtin signature (from compiler diagnostic): both pointers are to a 4xint
// vector, src in addrspace(1), dst in addrspace(3), then imm offset + cpol.
// Pointers are built from integers: low 32 bits of a generic shared address
// are the LDS byte offset.
// ---------------------------------------------------------------------------
typedef int gv4i __attribute__((vector_size(16)));
typedef __attribute__((address_space(1))) gv4i* gv4i_glob;
typedef __attribute__((address_space(3))) gv4i* gv4i_lds;

#if __has_builtin(__builtin_amdgcn_global_load_async_to_lds_b128)
#define ASYNC_COPY_B128(g, s)                                                  \
  __builtin_amdgcn_global_load_async_to_lds_b128(                              \
      (gv4i_glob)(unsigned long long)(g),                                      \
      (gv4i_lds)(unsigned)(unsigned long long)(s), 0, 0)
#else
#define ASYNC_COPY_B128(g, s)                                                  \
  asm volatile("global_load_async_to_lds_b128 %0, %1, off"                     \
               :: "v"((unsigned)(unsigned long long)(s)),                      \
                  "v"((unsigned long long)(g))                                 \
               : "memory")
#endif

#if __has_builtin(__builtin_amdgcn_s_wait_asynccnt)
#define WAIT_ASYNC(n) __builtin_amdgcn_s_wait_asynccnt(n)
#else
#define WAIT_ASYNC(n) asm volatile("s_wait_asynccnt %0" :: "n"(n) : "memory")
#endif

// ---------------------------------------------------------------------------
// LDS fragment loaders matching CDNA5 WMMA VGPR layouts (wave32).
// A (16x32 bf16): lanes 0-15 M=lane, K-halves 0..7 / 16..23;
//                 lanes16-31 M=lane-16, K-halves 8..15 / 24..31.
// B (32x16 bf16): lanes 0-15 N=lane, K-halves 0..7 / 16..23; lanes16-31 N=lane-16, K-halves 8..15 / 24..31.
// ---------------------------------------------------------------------------
__device__ __forceinline__ v16bf lds_frag_a(const unsigned char* base, int row0, int lane) {
  const unsigned char* p = base + (row0 + (lane & 15)) * ROWB + ((lane >> 4) << 4);
  v8bf lo = *(const v8bf*)p;            // K = kh .. kh+7
  v8bf hi = *(const v8bf*)(p + 32);     // K = kh+16 .. kh+23
  return __builtin_shufflevector(lo, hi, 0,1,2,3,4,5,6,7,8,9,10,11,12,13,14,15);
}

__device__ __forceinline__ v16bf lds_frag_b(const unsigned char* base, int col0, int lane) {
  const unsigned char* p = base + (col0 + (lane & 15)) * ROWB + ((lane >> 4) << 4);
  v8bf lo = *(const v8bf*)p;            // K = kb .. kb+7
  v8bf hi = *(const v8bf*)(p + 32);     // K = kb+16 .. kb+23
  return __builtin_shufflevector(lo, hi, 0,1,2,3,4,5,6,7,8,9,10,11,12,13,14,15);
}

__device__ __forceinline__ v8f wmma_bf16(v16bf a, v16bf b, v8f c) {
  return __builtin_amdgcn_wmma_f32_16x16x32_bf16(false, a, false, b,
                                                 (short)0, c, false, false);
}

// ---------------------------------------------------------------------------
// fp32 -> bf16 conversion, 4-wide
// ---------------------------------------------------------------------------
__global__ __launch_bounds__(256) void cvt_f32_bf16_4(
    const float* __restrict__ s, __bf16* __restrict__ d, int n4) {
  int i = blockIdx.x * 256 + threadIdx.x;
  if (i < n4) {
    const float4 v = ((const float4*)s)[i];
    v4bf o;
    o.x = (__bf16)v.x; o.y = (__bf16)v.y; o.z = (__bf16)v.z; o.w = (__bf16)v.w;
    ((v4bf*)d)[i] = o;
  }
}

// ---------------------------------------------------------------------------
// GEMM 1 (fused): gu = x @ w_gate_up^T, SwiGLU, write x_ffn in (B, L, INTER).
// A: (2048, 1024) bf16, W: (5632, 1024) bf16 (gate rows 0..2815, up 2816..).
// Block = 8 waves (2 M x 4 N): 128 rows x 128 features, gate AND up.
// Async-LDS double buffer: per stage A(8K)+Bg(8K)+Bu(8K) = 1536 x 16B chunks
// = 6 async B128 copies per thread.
// ---------------------------------------------------------------------------
__global__ __launch_bounds__(256) void gemm_gateup_swiglu(
    const __bf16* __restrict__ A, const __bf16* __restrict__ W,
    float* __restrict__ xf) {
  __shared__ __align__(16) unsigned char lds[2][STAGEB];
  const int tid  = threadIdx.x;
  const int lane = tid & 31;
  const int wave = tid >> 5;
  const int wm = wave >> 2;                   // 0..1
  const int wn = wave & 3;                    // 0..3
  const int m0b = blockIdx.y * 128;
  const int f0b = blockIdx.x * 128;

  v8f accg[4][2] = {};
  v8f accu[4][2] = {};

  auto stage = [&](int buf, int k0) {
#pragma unroll
    for (int j = 0; j < 6; ++j) {
      const int id   = tid + (j << 8);        // 0..1535
      const int ch   = id & 3;                // 16B chunk in 64B row
      const int r    = (id >> 2) & 127;       // row in its tile
      const int sect = id >> 9;               // 0=A, 1=Bg, 2=Bu
      const __bf16* src;
      if (sect == 0)      src = A + (size_t)(m0b + r) * DD + k0 + ch * 8;
      else if (sect == 1) src = W + (size_t)(f0b + r) * DD + k0 + ch * 8;
      else                src = W + (size_t)(f0b + r + INTER) * DD + k0 + ch * 8;
      unsigned char* dst = &lds[buf][sect * TILE_A + r * ROWB + ch * 16];
      ASYNC_COPY_B128(src, dst);
      if (j == 0) __builtin_prefetch(src + 64, 0, 1);   // pull K+64 into L2
    }
  };

  stage(0, 0);
  int buf = 0;
  for (int k0 = 0; k0 < DD; k0 += 32) {
    if (k0 + 32 < DD) {
      stage(buf ^ 1, k0 + 32);
      WAIT_ASYNC(6);          // in-order: current buffer's 6 copies complete
    } else {
      WAIT_ASYNC(0);
    }
    __syncthreads();

    const unsigned char* bA  = &lds[buf][0];
    const unsigned char* bBg = &lds[buf][TILE_A];
    const unsigned char* bBu = &lds[buf][2 * TILE_A];

    v16bf afr[4];
#pragma unroll
    for (int mi = 0; mi < 4; ++mi)
      afr[mi] = lds_frag_a(bA, wm * 64 + mi * 16, lane);
#pragma unroll
    for (int ni = 0; ni < 2; ++ni) {
      v16bf bg = lds_frag_b(bBg, wn * 32 + ni * 16, lane);
      v16bf bu = lds_frag_b(bBu, wn * 32 + ni * 16, lane);
#pragma unroll
      for (int mi = 0; mi < 4; ++mi) {
        accg[mi][ni] = wmma_bf16(afr[mi], bg, accg[mi][ni]);
        accu[mi][ni] = wmma_bf16(afr[mi], bu, accu[mi][ni]);
      }
    }
    __syncthreads();
    buf ^= 1;
  }

  // Epilogue: silu(gate) * up -> xf[(b*L+pix)*INTER + f]
#pragma unroll
  for (int mi = 0; mi < 4; ++mi) {
#pragma unroll
    for (int ni = 0; ni < 2; ++ni) {
      const int f   = f0b + wn * 32 + ni * 16 + (lane & 15);
      const int mrb = m0b + wm * 64 + mi * 16 + ((lane >> 4) << 3);
      v8f g = accg[mi][ni];
      v8f u = accu[mi][ni];
#pragma unroll
      for (int r = 0; r < 8; ++r) {
        float gv = g[r];
        float sv = gv / (1.0f + __expf(-gv));
        xf[(size_t)(mrb + r) * INTER + f] = sv * u[r];
      }
    }
  }
}

// ---------------------------------------------------------------------------
// Metric conv + offsets: one block per (b, pixel). Reduce 2816 ch x 9 taps
// into 7 outputs, then fused tanh/affine epilogue writing off_y/off_x.
// ---------------------------------------------------------------------------
__global__ __launch_bounds__(256) void metric_offsets_kernel(
    const float* __restrict__ xf, const float* __restrict__ wmet,
    const float* __restrict__ bmet,
    float* __restrict__ offy, float* __restrict__ offx) {
  const int b = blockIdx.y;
  const int pix = blockIdx.x;
  const int y = pix >> 5, x = pix & 31;
  const float* xb = xf + (size_t)b * LL * INTER;

  float acc[7] = {0.f, 0.f, 0.f, 0.f, 0.f, 0.f, 0.f};
  for (int c = threadIdx.x; c < INTER; c += 256) {
    float xn[9];
#pragma unroll
    for (int t = 0; t < 9; ++t) {
      const int yy = y + (t / 3) - 1;
      const int xx = x + (t % 3) - 1;
      const bool ok = (yy >= 0) & (yy < SS) & (xx >= 0) & (xx < SS);
      xn[t] = ok ? xb[(size_t)(yy * SS + xx) * INTER + c] : 0.f;
    }
#pragma unroll
    for (int o = 0; o < 7; ++o) {
      const float* wp = wmet + ((size_t)o * INTER + c) * 9;
      float s = 0.f;
#pragma unroll
      for (int t = 0; t < 9; ++t) s += xn[t] * wp[t];
      acc[o] += s;
    }
  }

  __shared__ float red[7][256];
#pragma unroll
  for (int o = 0; o < 7; ++o) red[o][threadIdx.x] = acc[o];
  __syncthreads();
  for (int s = 128; s > 0; s >>= 1) {
    if (threadIdx.x < (unsigned)s) {
#pragma unroll
      for (int o = 0; o < 7; ++o)
        red[o][threadIdx.x] += red[o][threadIdx.x + s];
    }
    __syncthreads();
  }

  if (threadIdx.x == 0) {
    float p[7];
#pragma unroll
    for (int o = 0; o < 7; ++o) p[o] = red[o][0] + bmet[o];
    const float scale = tanhf(p[6]);
#pragma unroll
    for (int k = 0; k < 9; ++k) {
      const float py = (float)((k / 3) - 1);
      const float px = (float)((k % 3) - 1);
      offy[((size_t)b * 9 + k) * LL + pix] = scale * (p[0] * py + p[1] * px) + p[4];
      offx[((size_t)b * 9 + k) * LL + pix] = scale * (p[2] * py + p[3] * px) + p[5];
    }
  }
}

// ---------------------------------------------------------------------------
// Deformable depthwise conv: thread-per-channel (coalesced gathers).
// Writes grid_out directly as bf16 A-matrix for GEMM2, (B*L, INTER) row-major.
// ---------------------------------------------------------------------------
__device__ __forceinline__ float dsample(const float* __restrict__ xb,
                                         int yy, int xx, int c) {
  const bool ok = (yy >= 0) & (yy < SS) & (xx >= 0) & (xx < SS);
  const int cy = min(max(yy, 0), SS - 1);
  const int cx = min(max(xx, 0), SS - 1);
  const float v = xb[(size_t)(cy * SS + cx) * INTER + c];
  return ok ? v : 0.f;
}

__global__ __launch_bounds__(256) void deform_dw_kernel(
    const float* __restrict__ xf, const float* __restrict__ offy,
    const float* __restrict__ offx, const float* __restrict__ cw,
    const float* __restrict__ cb, __bf16* __restrict__ gout) {
  const int b   = blockIdx.z;
  const int pix = blockIdx.y;
  const int c   = blockIdx.x * 256 + threadIdx.x;   // INTER = 11*256 exactly
  const int y = pix >> 5, x = pix & 31;
  const float* xb = xf + (size_t)b * LL * INTER;

  float acc = cb[c];
  const float* cwc = cw + (size_t)c * 9;
#pragma unroll
  for (int k = 0; k < 9; ++k) {
    const float py = (float)((k / 3) - 1);
    const float px = (float)((k % 3) - 1);
    const float oy = offy[((size_t)b * 9 + k) * LL + pix];   // uniform
    const float ox = offx[((size_t)b * 9 + k) * LL + pix];   // uniform
    const float pos_y = (float)y + py + oy;
    const float pos_x = (float)x + px + ox;
    const float y0f = floorf(pos_y);
    const float x0f = floorf(pos_x);
    const float ty = pos_y - y0f;
    const float tx = pos_x - x0f;
    const int y0 = (int)y0f, x0 = (int)x0f;
    const float v00 = dsample(xb, y0,     x0,     c);
    const float v01 = dsample(xb, y0,     x0 + 1, c);
    const float v10 = dsample(xb, y0 + 1, x0,     c);
    const float v11 = dsample(xb, y0 + 1, x0 + 1, c);
    const float bil = (1.f - ty) * ((1.f - tx) * v00 + tx * v01)
                    +        ty  * ((1.f - tx) * v10 + tx * v11);
    acc += cwc[k] * bil;
  }
  gout[((size_t)b * LL + pix) * INTER + c] = (__bf16)acc;
}

// ---------------------------------------------------------------------------
// GEMM 2: out = grid_out @ w_down^T.
// A: (2048, 2816) bf16, W: (1024, 2816) bf16, out: (2048, 1024) fp32.
// Block = 8 waves (2 M x 4 N): 128 x 256 tile; wave 64x64 (4x4 subtiles).
// Async-LDS double buffer: A(8K)+B(16K) = 1536 x 16B chunks = 6 per thread.
// ---------------------------------------------------------------------------
__global__ __launch_bounds__(256) void gemm_down(
    const __bf16* __restrict__ A, const __bf16* __restrict__ W,
    float* __restrict__ out) {
  __shared__ __align__(16) unsigned char lds[2][STAGEB];
  const int tid  = threadIdx.x;
  const int lane = tid & 31;
  const int wave = tid >> 5;
  const int wm = wave >> 2;                   // 0..1
  const int wn = wave & 3;                    // 0..3
  const int m0b = blockIdx.y * 128;
  const int n0b = blockIdx.x * 256;

  v8f acc[4][4] = {};

  auto stage = [&](int buf, int k0) {
#pragma unroll
    for (int j = 0; j < 6; ++j) {
      const int id = tid + (j << 8);          // 0..1535
      const int ch = id & 3;
      const __bf16* src;
      unsigned char* dst;
      if (id < 512) {                          // A tile: 128 rows
        const int r = id >> 2;
        src = A + (size_t)(m0b + r) * INTER + k0 + ch * 8;
        dst = &lds[buf][r * ROWB + ch * 16];
      } else {                                 // B tile: 256 cols
        const int cidx = (id >> 2) - 128;
        src = W + (size_t)(n0b + cidx) * INTER + k0 + ch * 8;
        dst = &lds[buf][TILE_A + cidx * ROWB + ch * 16];
      }
      ASYNC_COPY_B128(src, dst);
      if (j == 0) __builtin_prefetch(src + 64, 0, 1);
    }
  };

  stage(0, 0);
  int buf = 0;
  for (int k0 = 0; k0 < INTER; k0 += 32) {
    if (k0 + 32 < INTER) {
      stage(buf ^ 1, k0 + 32);
      WAIT_ASYNC(6);
    } else {
      WAIT_ASYNC(0);
    }
    __syncthreads();

    const unsigned char* bA = &lds[buf][0];
    const unsigned char* bB = &lds[buf][TILE_A];

    v16bf afr[4], bfr[4];
#pragma unroll
    for (int mi = 0; mi < 4; ++mi)
      afr[mi] = lds_frag_a(bA, wm * 64 + mi * 16, lane);
#pragma unroll
    for (int ni = 0; ni < 4; ++ni)
      bfr[ni] = lds_frag_b(bB, wn * 64 + ni * 16, lane);
#pragma unroll
    for (int mi = 0; mi < 4; ++mi)
#pragma unroll
      for (int ni = 0; ni < 4; ++ni)
        acc[mi][ni] = wmma_bf16(afr[mi], bfr[ni], acc[mi][ni]);

    __syncthreads();
    buf ^= 1;
  }

#pragma unroll
  for (int mi = 0; mi < 4; ++mi) {
#pragma unroll
    for (int ni = 0; ni < 4; ++ni) {
      const int n   = n0b + wn * 64 + ni * 16 + (lane & 15);
      const int mrb = m0b + wm * 64 + mi * 16 + ((lane >> 4) << 3);
      v8f a = acc[mi][ni];
#pragma unroll
      for (int r = 0; r < 8; ++r)
        out[(size_t)(mrb + r) * DD + n] = a[r];
    }
  }
}

// ---------------------------------------------------------------------------
// Launch
// ---------------------------------------------------------------------------
extern "C" void kernel_launch(void* const* d_in, const int* in_sizes, int n_in,
                              void* d_out, int out_size, void* d_ws, size_t ws_size,
                              hipStream_t stream) {
  (void)in_sizes; (void)n_in; (void)out_size; (void)ws_size;
  const float* x    = (const float*)d_in[0];   // (B, L, D)
  const float* wgu  = (const float*)d_in[1];   // (2*INTER, D)
  const float* wmet = (const float*)d_in[2];   // (7, INTER, 3, 3)
  const float* bmet = (const float*)d_in[3];   // (7,)
  const float* cw   = (const float*)d_in[4];   // (INTER, 1, 3, 3)
  const float* cb   = (const float*)d_in[5];   // (INTER,)
  const float* wd   = (const float*)d_in[6];   // (D, INTER)
  float* out = (float*)d_out;                  // (B, L, D) fp32

  char* ws = (char*)d_ws;
  const size_t nA   = (size_t)BB * LL * DD;
  const size_t nWgu = (size_t)2 * INTER * DD;
  const size_t nWd  = (size_t)DD * INTER;
  const size_t nXF  = (size_t)BB * LL * INTER;
  const size_t nOff = (size_t)BB * 9 * LL;

  size_t off = 0;
  auto carve = [&](size_t bytes) { size_t r = off; off = (off + bytes + 255) & ~(size_t)255; return r; };
  __bf16* Abf  = (__bf16*)(ws + carve(nA * 2));
  __bf16* Wgub = (__bf16*)(ws + carve(nWgu * 2));
  __bf16* Wdb  = (__bf16*)(ws + carve(nWd * 2));
  float*  xf   = (float*) (ws + carve(nXF * 4));
  float*  offy = (float*) (ws + carve(nOff * 4));
  float*  offx = (float*) (ws + carve(nOff * 4));
  __bf16* gout = (__bf16*)(ws + carve(nXF * 2));

  // 1) fp32 -> bf16 conversions (all counts divisible by 4)
  cvt_f32_bf16_4<<<dim3((unsigned)(nA / 4 + 255) / 256),   256, 0, stream>>>(x,   Abf,  (int)(nA / 4));
  cvt_f32_bf16_4<<<dim3((unsigned)(nWgu / 4 + 255) / 256), 256, 0, stream>>>(wgu, Wgub, (int)(nWgu / 4));
  cvt_f32_bf16_4<<<dim3((unsigned)(nWd / 4 + 255) / 256),  256, 0, stream>>>(wd,  Wdb,  (int)(nWd / 4));

  // 2) gate/up GEMM + fused SwiGLU -> x_ffn (B, L, INTER) fp32
  gemm_gateup_swiglu<<<dim3(INTER / 128, (BB * LL) / 128), 256, 0, stream>>>(Abf, Wgub, xf);

  // 3) metric conv + offsets
  metric_offsets_kernel<<<dim3(LL, BB), 256, 0, stream>>>(xf, wmet, bmet, offy, offx);

  // 4) deformable depthwise -> grid_out bf16 (B*L, INTER)
  deform_dw_kernel<<<dim3(INTER / 256, LL, BB), 256, 0, stream>>>(xf, offy, offx, cw, cb, gout);

  // 5) down GEMM -> out fp32
  gemm_down<<<dim3(DD / 256, (BB * LL) / 128), 256, 0, stream>>>(gout, Wdb, out);
}
